// HypergraphAttentionConv_18090402251351
// MI455X (gfx1250) — matrix operand order repaired and checked by reading, not compiled, not run on hardware
//
#include <hip/hip_runtime.h>

// ---------------------------------------------------------------------------
// HypergraphAttentionConv for MI455X (gfx1250, wave32, WMMA)
//
// Fused algebra:  out = diag(1/(Dv+eps)) * ( alpha @ ( (alpha^T @ Xt) / (De+eps) ) )
// so the N x N H_norm matrix is never materialized.
// GEMMs: v_wmma_f32_16x16x32_bf16, double-buffered LDS tiles, A staged via
// async global->LDS (ASYNCcnt) when the toolchain exposes the builtins.
// ---------------------------------------------------------------------------

typedef __attribute__((ext_vector_type(16))) __bf16 v16bf;
typedef __attribute__((ext_vector_type(8)))  float  v8f;
typedef int v4i_vs __attribute__((vector_size(16)));              // matches builtin proto
typedef __attribute__((address_space(1))) v4i_vs* gv4i_ptr;       // global int4*
typedef __attribute__((address_space(3))) v4i_vs* lv4i_ptr;       // LDS int4*

#define NN    4096
#define EE    512
#define FIN   256
#define FOUTC 256
#define HIDC  64

#define EPSF      1e-9f
#define NEGINF    (-1000000000.0f)
#define NEGSLOPE  0.2f

#if __has_builtin(__builtin_amdgcn_global_load_async_to_lds_b128) && \
    __has_builtin(__builtin_amdgcn_s_wait_asynccnt)
#define USE_ASYNC_LDS 1
#else
#define USE_ASYNC_LDS 0
#endif

// --------------------------------------------------------------------------
// f32 -> bf16 elementwise convert
// --------------------------------------------------------------------------
__global__ void k_f32_to_bf16(const float* __restrict__ in, __bf16* __restrict__ out, int n) {
    int i = blockIdx.x * blockDim.x + threadIdx.x;
    int stride = gridDim.x * blockDim.x;
    for (; i < n; i += stride) out[i] = (__bf16)in[i];
}

// --------------------------------------------------------------------------
// Transpose H [NN, EE] f32  ->  Ht [EE, NN] bf16   (tiled through LDS)
// --------------------------------------------------------------------------
__global__ void k_transpose_h(const float* __restrict__ H, __bf16* __restrict__ Ht) {
    __shared__ float tile[32][33];
    int e0 = blockIdx.x * 32;
    int n0 = blockIdx.y * 32;
    int tx = threadIdx.x;   // 0..31
    int ty = threadIdx.y;   // 0..7
    #pragma unroll
    for (int j = 0; j < 4; ++j) {
        int r = ty + j * 8;
        tile[r][tx] = H[(size_t)(n0 + r) * EE + e0 + tx];
    }
    __syncthreads();
    #pragma unroll
    for (int j = 0; j < 4; ++j) {
        int r = ty + j * 8;
        Ht[(size_t)(e0 + r) * NN + n0 + tx] = (__bf16)tile[tx][r];
    }
}

// --------------------------------------------------------------------------
// edge_sizes[e] = sum_n Ht[e, n]   (raw sum; +eps folded into GEMM epilogue)
// --------------------------------------------------------------------------
__global__ void k_edge_sizes(const __bf16* __restrict__ Ht, float* __restrict__ es) {
    __shared__ float red[256];
    int e = blockIdx.x;
    int t = threadIdx.x;
    const __bf16* row = Ht + (size_t)e * NN;
    float s = 0.f;
    for (int i = t; i < NN; i += 256) s += (float)row[i];
    red[t] = s;
    __syncthreads();
    for (int w = 128; w > 0; w >>= 1) {
        if (t < w) red[t] += red[t + w];
        __syncthreads();
    }
    if (t == 0) es[e] = red[0];
}

// --------------------------------------------------------------------------
// Tiled bf16 WMMA GEMM:  C[M,N] = A[M,K] @ B[K,N]
// A,B bf16 row-major (lda=K, ldb=N).  Optional epilogue:
//   val /= (rowDiv[row] + div_eps)   if rowDiv != nullptr
//   C (f32) and/or Cbf (bf16) written if non-null.
// Block tile 128(M) x 64(N), 8 waves of 32, wave tile 32x32 (2x2 wmma).
// Double-buffered LDS; A tiles DMA'd with global_load_async_to_lds_b128
// when available. B staged transposed ([col][k], stride 40) so fragments
// are contiguous 32B runs (2 x ds_load_b128, conflict-free banks).
// Requires M%128==0, N%64==0, K%32==0 (true for all six GEMMs here).
// --------------------------------------------------------------------------
#define APAD 40   // A row stride in bf16 (16B aligned, bank-spread)
#define BPAD 40   // Bt row stride in bf16

__global__ __launch_bounds__(256)
void k_gemm_bf16(const __bf16* __restrict__ A, const __bf16* __restrict__ B,
                 float* __restrict__ C, __bf16* __restrict__ Cbf,
                 const float* __restrict__ rowDiv, float div_eps,
                 int M, int N, int K) {
    __shared__ __align__(16) __bf16 lA[2][128 * APAD];   // 2 x 10 KB
    __shared__ __align__(16) __bf16 lBt[2][64 * BPAD];   // 2 x  5 KB, [col][k]

    const int tid  = threadIdx.x;
    const int lane = tid & 31;
    const int wave = tid >> 5;
    const int wm = wave >> 1;          // 0..3  (M direction, 32 each)
    const int wn = wave & 1;           // 0..1  (N direction, 32 each)
    const int lm = lane & 15;
    const int lh = lane >> 4;

    const int mBase = blockIdx.y * 128;
    const int nBase = blockIdx.x * 64;

    // staging coordinates
    const int arow0 = tid >> 2,          ac0 = (tid & 3) * 8;         // A part 0
    const int arow1 = (tid + 256) >> 2,  ac1 = ((tid + 256) & 3) * 8; // A part 1
    const int bkrow = tid >> 3,          bc0 = (tid & 7) * 8;         // B row, col group

    v8f acc[2][2];
    #pragma unroll
    for (int f = 0; f < 2; ++f)
        #pragma unroll
        for (int g = 0; g < 2; ++g)
            #pragma unroll
            for (int r = 0; r < 8; ++r) acc[f][g][r] = 0.0f;

    union Q8 { uint4 q; __bf16 h[8]; };

    // ---- tile staging: A (async if available) + B transposed (sync) ----
    auto stage = [&](int buf, int k0) {
#if USE_ASYNC_LDS
        __builtin_amdgcn_global_load_async_to_lds_b128(
            (gv4i_ptr)(A + (size_t)(mBase + arow0) * K + k0 + ac0),
            (lv4i_ptr)(&lA[buf][arow0 * APAD + ac0]), 0, 0);
        __builtin_amdgcn_global_load_async_to_lds_b128(
            (gv4i_ptr)(A + (size_t)(mBase + arow1) * K + k0 + ac1),
            (lv4i_ptr)(&lA[buf][arow1 * APAD + ac1]), 0, 0);
#else
        {
            const uint4* pa0 = reinterpret_cast<const uint4*>(A + (size_t)(mBase + arow0) * K + k0 + ac0);
            const uint4* pa1 = reinterpret_cast<const uint4*>(A + (size_t)(mBase + arow1) * K + k0 + ac1);
            uint4 va0 = *pa0, va1 = *pa1;
            *reinterpret_cast<uint4*>(&lA[buf][arow0 * APAD + ac0]) = va0;
            *reinterpret_cast<uint4*>(&lA[buf][arow1 * APAD + ac1]) = va1;
        }
#endif
        // B: load 8 consecutive cols of row (k0+bkrow), scatter transposed
        Q8 vb;
        vb.q = *reinterpret_cast<const uint4*>(B + (size_t)(k0 + bkrow) * N + nBase + bc0);
        #pragma unroll
        for (int j = 0; j < 8; ++j)
            lBt[buf][(bc0 + j) * BPAD + bkrow] = vb.h[j];

        if (k0 + 32 < K) { // pull next K tile toward L2 (global_prefetch_b8)
            __builtin_prefetch(A + (size_t)(mBase + arow0) * K + k0 + 32, 0, 0);
            __builtin_prefetch(B + (size_t)(k0 + 32 + bkrow) * N + nBase, 0, 0);
        }
    };

    stage(0, 0);   // prologue

    const int ksteps = K >> 5;
    for (int kt = 0; kt < ksteps; ++kt) {
        const int cur = kt & 1;
#if USE_ASYNC_LDS
        __builtin_amdgcn_s_wait_asynccnt(0);
#endif
        __syncthreads();                       // tile `cur` fully resident

        if (kt + 1 < ksteps) stage(1 - cur, (kt + 1) << 5);  // overlap next DMA

        // ---- build fragments (all contiguous 16B runs in LDS) ----
        v16bf fa[2], fb[2];
        #pragma unroll
        for (int f = 0; f < 2; ++f) {
            int row = wm * 32 + f * 16 + lm;
            const __bf16* p = &lA[cur][row * APAD + 16 * lh];
            #pragma unroll
            for (int j = 0; j < 8; ++j) fa[f][j] = p[j];          // K = 16*lh + 0..7
            #pragma unroll
            for (int j = 0; j < 8; ++j) fa[f][8 + j] = p[8 + j];  // K = 16*lh + 8..15
        }
        #pragma unroll
        for (int g = 0; g < 2; ++g) {
            int col = wn * 32 + g * 16 + lm;
            const __bf16* p = &lBt[cur][col * BPAD + 16 * lh];
            #pragma unroll
            for (int i = 0; i < 16; ++i) fb[g][i] = p[i];         // K = 16*lh + 0..15
        }

        #pragma unroll
        for (int f = 0; f < 2; ++f)
            #pragma unroll
            for (int g = 0; g < 2; ++g)
                acc[f][g] = __builtin_amdgcn_wmma_f32_16x16x32_bf16(
                    false, fa[f], false, fb[g], (short)0, acc[f][g], false, false);
    }

    // ---- epilogue ----
    #pragma unroll
    for (int f = 0; f < 2; ++f) {
        #pragma unroll
        for (int g = 0; g < 2; ++g) {
            #pragma unroll
            for (int r = 0; r < 8; ++r) {
                int gr = mBase + wm * 32 + f * 16 + lh * 8 + r;
                int gc = nBase + wn * 32 + g * 16 + lm;
                float val = acc[f][g][r];
                if (rowDiv) val /= (rowDiv[gr] + div_eps);
                size_t idx = (size_t)gr * N + gc;
                if (C)   C[idx]   = val;
                if (Cbf) Cbf[idx] = (__bf16)val;
            }
        }
    }
}

// --------------------------------------------------------------------------
// Attention score: sT[e,n] = mask ? leaky_relu(s_node[n]+s_edge[e]) . attn_v
//                           : NEG_INF
// --------------------------------------------------------------------------
__global__ void k_score(const float* __restrict__ s_node, const float* __restrict__ s_edge,
                        const float* __restrict__ attn_v, const __bf16* __restrict__ Ht,
                        float* __restrict__ sT) {
    __shared__ float sn[16][65];
    __shared__ float se[16][65];
    __shared__ float av[HIDC];
    int e0 = blockIdx.x * 16;
    int n0 = blockIdx.y * 16;
    int tx = threadIdx.x, ty = threadIdx.y;
    int tid = ty * 16 + tx;
    #pragma unroll
    for (int p = 0; p < 4; ++p) {
        int l = tid + p * 256;          // 0..1023 over 16x64
        int row = l >> 6, col = l & 63;
        sn[row][col] = s_node[(size_t)(n0 + row) * HIDC + col];
        se[row][col] = s_edge[(size_t)(e0 + row) * HIDC + col];
    }
    if (tid < HIDC) av[tid] = attn_v[tid];
    __syncthreads();

    int n = n0 + tx;
    int e = e0 + ty;
    float acc = 0.f;
    #pragma unroll
    for (int h = 0; h < HIDC; ++h) {
        float v = sn[tx][h] + se[ty][h];
        float lr = (v > 0.f) ? v : NEGSLOPE * v;
        acc += lr * av[h];
    }
    bool msk = ((float)Ht[(size_t)e * NN + n]) > 0.f;
    sT[(size_t)e * NN + n] = msk ? acc : NEGINF;
}

// --------------------------------------------------------------------------
// Per-edge (column) softmax over N, masked.  One block per edge.
// --------------------------------------------------------------------------
__global__ void k_softmax(const float* __restrict__ sT, const __bf16* __restrict__ Ht,
                          float* __restrict__ alpha_f32, __bf16* __restrict__ alphaT_bf,
                          float* __restrict__ De) {
    __shared__ float red[256];
    int e = blockIdx.x;
    int t = threadIdx.x;
    const float* row = sT + (size_t)e * NN;
    const __bf16* mrow = Ht + (size_t)e * NN;

    // pass 1: max
    float m = -3.4e38f;
    for (int i = t; i < NN; i += 256) m = fmaxf(m, row[i]);
    red[t] = m;
    __syncthreads();
    for (int w = 128; w > 0; w >>= 1) { if (t < w) red[t] = fmaxf(red[t], red[t + w]); __syncthreads(); }
    float mx = red[0];
    __syncthreads();

    // pass 2: Z (all terms) and masked sum
    float zall = 0.f, zmask = 0.f;
    for (int i = t; i < NN; i += 256) {
        float ex = __expf(row[i] - mx);
        zall += ex;
        if (((float)mrow[i]) > 0.f) zmask += ex;
    }
    red[t] = zall;
    __syncthreads();
    for (int w = 128; w > 0; w >>= 1) { if (t < w) red[t] += red[t + w]; __syncthreads(); }
    float Z = red[0];
    __syncthreads();
    red[t] = zmask;
    __syncthreads();
    for (int w = 128; w > 0; w >>= 1) { if (t < w) red[t] += red[t + w]; __syncthreads(); }
    float SM = red[0];
    float invZ = 1.0f / Z;

    // pass 3: write alpha
    for (int i = t; i < NN; i += 256) {
        bool msk = ((float)mrow[i]) > 0.f;
        float a = msk ? __expf(row[i] - mx) * invZ : 0.f;
        alpha_f32[(size_t)i * EE + e] = a;
        alphaT_bf[(size_t)e * NN + i] = (__bf16)a;
    }
    if (t == 0) De[e] = SM * invZ;   // raw; +eps folded into GEMM epilogue
}

// --------------------------------------------------------------------------
// Dv[n] = sum_e alpha[n,e] (deterministic row reduction, no FP atomics)
// Also emits the row-major bf16 copy of alpha for the final GEMM.
// --------------------------------------------------------------------------
__global__ void k_dv(const float* __restrict__ alpha_f32, float* __restrict__ Dv,
                     __bf16* __restrict__ alpha_bf) {
    __shared__ float red[128];
    int n = blockIdx.x;
    int t = threadIdx.x;
    const float* row = alpha_f32 + (size_t)n * EE;
    float s = 0.f;
    #pragma unroll
    for (int p = 0; p < 4; ++p) {
        int i = t + p * 128;
        float a = row[i];
        s += a;
        alpha_bf[(size_t)n * EE + i] = (__bf16)a;
    }
    red[t] = s;
    __syncthreads();
    for (int w = 64; w > 0; w >>= 1) { if (t < w) red[t] += red[t + w]; __syncthreads(); }
    if (t == 0) Dv[n] = red[0];      // raw; +eps folded into GEMM epilogue
}

// --------------------------------------------------------------------------
extern "C" void kernel_launch(void* const* d_in, const int* in_sizes, int n_in,
                              void* d_out, int out_size, void* d_ws, size_t ws_size,
                              hipStream_t stream) {
    const float* X      = (const float*)d_in[0];   // [NN, FIN]
    const float* H      = (const float*)d_in[1];   // [NN, EE]
    const float* W      = (const float*)d_in[2];   // [FIN, FOUT]
    const float* a_node = (const float*)d_in[3];   // [FOUT, HID]
    const float* a_edge = (const float*)d_in[4];   // [FOUT, HID]
    const float* attn_v = (const float*)d_in[5];   // [HID, 1]
    float* out = (float*)d_out;                    // [NN, FOUT]

    // ---- workspace carve-up (256B aligned) ----
    char* wp = (char*)d_ws;
    auto carve = [&](size_t bytes) {
        void* p = (void*)wp;
        wp += (bytes + 255) & ~(size_t)255;
        return p;
    };
    __bf16* Xbf   = (__bf16*)carve((size_t)NN * FIN * 2);
    __bf16* Wbf   = (__bf16*)carve((size_t)FIN * FOUTC * 2);
    __bf16* anb   = (__bf16*)carve((size_t)FOUTC * HIDC * 2);
    __bf16* aeb   = (__bf16*)carve((size_t)FOUTC * HIDC * 2);
    __bf16* Htb   = (__bf16*)carve((size_t)EE * NN * 2);
    float*  es    = (float*) carve((size_t)EE * 4);
    __bf16* Xtb   = (__bf16*)carve((size_t)NN * FOUTC * 2);
    __bf16* Heb   = (__bf16*)carve((size_t)EE * FOUTC * 2);
    float*  snode = (float*) carve((size_t)NN * HIDC * 4);
    float*  sedge = (float*) carve((size_t)EE * HIDC * 4);
    float*  sT    = (float*) carve((size_t)EE * NN * 4);
    __bf16* aTbf  = (__bf16*)carve((size_t)EE * NN * 2);
    float*  aF32  = (float*) carve((size_t)NN * EE * 4);
    __bf16* aBf   = (__bf16*)carve((size_t)NN * EE * 2);
    float*  De    = (float*) carve((size_t)EE * 4);
    float*  Dv    = (float*) carve((size_t)NN * 4);
    __bf16* M2bf  = (__bf16*)carve((size_t)EE * FOUTC * 2);

    // ---- 1) f32 -> bf16 converts ----
    k_f32_to_bf16<<<dim3(256), dim3(256), 0, stream>>>(X, Xbf, NN * FIN);
    k_f32_to_bf16<<<dim3(64),  dim3(256), 0, stream>>>(W, Wbf, FIN * FOUTC);
    k_f32_to_bf16<<<dim3(16),  dim3(256), 0, stream>>>(a_node, anb, FOUTC * HIDC);
    k_f32_to_bf16<<<dim3(16),  dim3(256), 0, stream>>>(a_edge, aeb, FOUTC * HIDC);

    // ---- 2) Ht (transpose of H) + edge sizes ----
    k_transpose_h<<<dim3(EE / 32, NN / 32), dim3(32, 8), 0, stream>>>(H, Htb);
    k_edge_sizes<<<dim3(EE), dim3(256), 0, stream>>>(Htb, es);

    // ---- 3) Xt = X @ W  (bf16 out for downstream GEMMs) ----
    k_gemm_bf16<<<dim3(FOUTC / 64, NN / 128), dim3(256), 0, stream>>>(
        Xbf, Wbf, nullptr, Xtb, nullptr, 0.f, NN, FOUTC, FIN);

    // ---- 4) s_node = Xt @ a_node ----
    k_gemm_bf16<<<dim3(HIDC / 64, NN / 128), dim3(256), 0, stream>>>(
        Xtb, anb, snode, nullptr, nullptr, 0.f, NN, HIDC, FOUTC);

    // ---- 5) He = (H^T @ Xt) / (edge_sizes + eps) ----
    k_gemm_bf16<<<dim3(FOUTC / 64, EE / 128), dim3(256), 0, stream>>>(
        Htb, Xtb, nullptr, Heb, es, EPSF, EE, FOUTC, NN);

    // ---- 6) s_edge = He @ a_edge ----
    k_gemm_bf16<<<dim3(HIDC / 64, EE / 128), dim3(256), 0, stream>>>(
        Heb, aeb, sedge, nullptr, nullptr, 0.f, EE, HIDC, FOUTC);

    // ---- 7) masked attention scores sT[e,n] ----
    k_score<<<dim3(EE / 16, NN / 16), dim3(16, 16), 0, stream>>>(
        snode, sedge, attn_v, Htb, sT);

    // ---- 8) per-edge softmax -> alpha, alpha^T, De ----
    k_softmax<<<dim3(EE), dim3(256), 0, stream>>>(sT, Htb, aF32, aTbf, De);

    // ---- 9) Dv row sums + bf16 alpha ----
    k_dv<<<dim3(NN), dim3(128), 0, stream>>>(aF32, Dv, aBf);

    // ---- 10) M2 = (alpha^T @ Xt) / (De + eps) ----
    k_gemm_bf16<<<dim3(FOUTC / 64, EE / 128), dim3(256), 0, stream>>>(
        aTbf, Xtb, nullptr, M2bf, De, EPSF, EE, FOUTC, NN);

    // ---- 11) out = (alpha @ M2) / (Dv + eps) ----
    k_gemm_bf16<<<dim3(FOUTC / 64, NN / 128), dim3(256), 0, stream>>>(
        aBf, M2bf, out, nullptr, Dv, EPSF, NN, FOUTC, EE);
}
